// CausalSelfAttention_67327907332361
// MI455X (gfx1250) — matrix operand order repaired and checked by reading, not compiled, not run on hardware
//
#include <hip/hip_runtime.h>
#include <hip/hip_bf16.h>
#include <stdint.h>

typedef __bf16 bf16_t;
typedef __attribute__((ext_vector_type(16))) __bf16 v16bf;
typedef __attribute__((ext_vector_type(8)))  __bf16 v8bf;
typedef __attribute__((ext_vector_type(8)))  float  v8f;

#define DEVINL __device__ __forceinline__

// Problem constants (from reference)
constexpr int BSZ    = 2;
constexpr int SEQ    = 2048;
constexpr int CDIM   = 2048;
constexpr int NH     = 16;
constexpr int NKV    = 4;
constexpr int HD     = 128;
constexpr int WIN    = 1024;
constexpr int SINKN  = 4;
constexpr float SCALE = 0.08838834764831845f; // 1/sqrt(128)

// ---------- helpers ----------
DEVINL bf16_t f2bf(float f) {
  uint32_t u = __float_as_uint(f);
  uint32_t r = u + 0x7fffu + ((u >> 16) & 1u);   // round-to-nearest-even
  uint16_t h = (uint16_t)(r >> 16);
  return __builtin_bit_cast(bf16_t, h);
}

DEVINL v16bf cat8(v8bf lo, v8bf hi) {
  return __builtin_shufflevector(lo, hi, 0,1,2,3,4,5,6,7,8,9,10,11,12,13,14,15);
}
DEVINL v8bf ld8(const bf16_t* p) { return *(const v8bf*)p; }

DEVINL v8f wmma_bf16(v16bf a, v16bf b, v8f c) {
  // D = A(16x32) * B(32x16) + C, fp32 accumulate
  return __builtin_amdgcn_wmma_f32_16x16x32_bf16(false, a, false, b, (short)0, c,
                                                 false, false);
}

// A fragment (16x32): lane m=L&15; hi=L>>4; K chunks at {hi*8, 16+hi*8}
DEVINL v16bf load_a(const bf16_t* row, int hi) {
  return cat8(ld8(row + hi * 8), ld8(row + 16 + hi * 8));
}
// B fragment (32x16, NT source row = output column): K chunks at {hi*8, 16+hi*8}
DEVINL v16bf load_b(const bf16_t* row, int hi) {
  return cat8(ld8(row + hi * 8), ld8(row + 16 + hi * 8));
}

// ---------- elementwise f32 -> bf16 (8 elements / thread / step) ----------
__global__ void convert_bf16_v8(const float* __restrict__ in,
                                bf16_t* __restrict__ out, long n8) {
  long i = (long)blockIdx.x * blockDim.x + threadIdx.x;
  long stride = (long)gridDim.x * blockDim.x;
  for (; i < n8; i += stride) {
    const float4* p = (const float4*)(in + i * 8);
    float4 a = p[0], b = p[1];
    v8bf o;
    o[0] = f2bf(a.x); o[1] = f2bf(a.y); o[2] = f2bf(a.z); o[3] = f2bf(a.w);
    o[4] = f2bf(b.x); o[5] = f2bf(b.y); o[6] = f2bf(b.z); o[7] = f2bf(b.w);
    *(v8bf*)(out + i * 8) = o;
  }
}

// ---------- GEMM: C[MxN] = A[MxK] * B[NxK]^T  (bf16 in, f32 out) ----------
// block = 256 threads (8 waves, 4 along M x 2 along N); block tile 256x128;
// wave tile 64x64 -> 16 WMMA per 8 fragment loads per K-step (K-step = 32).
__global__ __launch_bounds__(256) void gemm_nt(const bf16_t* __restrict__ A,
                                               const bf16_t* __restrict__ Bm,
                                               float* __restrict__ Cm,
                                               int M, int N, int K) {
  const int lane = threadIdx.x & 31;
  const int wid  = threadIdx.x >> 5;
  const int nlo  = lane & 15;
  const int hi   = lane >> 4;
  const int wm = wid & 3;   // 4 waves along M
  const int wn = wid >> 2;  // 2 waves along N
  const int m0 = blockIdx.y * 256 + wm * 64;
  const int n0 = blockIdx.x * 128 + wn * 64;

  v8f acc[4][4];
#pragma unroll
  for (int i = 0; i < 4; ++i)
#pragma unroll
    for (int j = 0; j < 4; ++j) acc[i][j] = (v8f){0.f,0.f,0.f,0.f,0.f,0.f,0.f,0.f};

  const bf16_t* arow = A + (long)(m0 + nlo) * K;
  const bf16_t* brow = Bm + (long)(n0 + nlo) * K;
  for (int k0 = 0; k0 < K; k0 += 32) {
    v16bf a[4], b[4];
#pragma unroll
    for (int i = 0; i < 4; ++i) a[i] = load_a(arow + (long)i * 16 * K + k0, hi);
#pragma unroll
    for (int j = 0; j < 4; ++j) b[j] = load_b(brow + (long)j * 16 * K + k0, hi);
#pragma unroll
    for (int i = 0; i < 4; ++i)
#pragma unroll
      for (int j = 0; j < 4; ++j) acc[i][j] = wmma_bf16(a[i], b[j], acc[i][j]);
  }

#pragma unroll
  for (int i = 0; i < 4; ++i)
#pragma unroll
    for (int j = 0; j < 4; ++j) {
      float* cp = Cm + (long)(m0 + i * 16 + 8 * hi) * N + n0 + j * 16 + nlo;
#pragma unroll
      for (int r = 0; r < 8; ++r) cp[(long)r * N] = acc[i][j][r];
    }
}

// ---------- RoPE (fp32 in, bf16 out); layout (B,T,nheads,128) ----------
__global__ void rope_convert(const float* __restrict__ in, bf16_t* __restrict__ out,
                             int nheads, long total) {
  long idx = (long)blockIdx.x * blockDim.x + threadIdx.x;
  if (idx >= total) return;
  int p   = (int)(idx & 63);
  long t1 = idx >> 6;
  int hh  = (int)(t1 % nheads);
  long t2 = t1 / nheads;
  int t   = (int)(t2 % SEQ);
  int bb  = (int)(t2 / SEQ);
  long base = (((long)bb * SEQ + t) * nheads + hh) * HD;
  float x1 = in[base + p], x2 = in[base + 64 + p];
  float inv = __powf(10000.0f, -(float)(2 * p) / 128.0f);
  float ang = (float)t * inv;
  float c, s;
  sincosf(ang, &s, &c);
  out[base + p]      = f2bf(x1 * c - x2 * s);
  out[base + 64 + p] = f2bf(x2 * c + x1 * s);
}

// ---------- V transpose: (B,T,NKV,HD) f32 -> (B,NKV,HD,T) bf16 ----------
__global__ __launch_bounds__(256) void transpose_v(const float* __restrict__ v,
                                                   bf16_t* __restrict__ vt) {
  __shared__ float tile[32][33];
  int bb = blockIdx.z / NKV, kv = blockIdx.z % NKV;
  int d0 = blockIdx.x * 32, t0 = blockIdx.y * 32;
  int x = threadIdx.x & 31, y = threadIdx.x >> 5;
  for (int yy = y; yy < 32; yy += 8)
    tile[yy][x] = v[(((long)bb * SEQ + t0 + yy) * NKV + kv) * HD + d0 + x];
  __syncthreads();
  for (int yy = y; yy < 32; yy += 8)
    vt[(((long)bb * NKV + kv) * HD + d0 + yy) * SEQ + t0 + x] = f2bf(tile[x][yy]);
}

// ---------- Flash attention: one wave per 16-row Q tile per (b,h) ----------
// q: (B,T,NH,HD) bf16 (rope'd), k: (B,T,NKV,HD) bf16 (rope'd),
// vt: (B,NKV,HD,T) bf16, y: (B,T,NH,HD) bf16
__global__ __launch_bounds__(256) void attn_fwd(const bf16_t* __restrict__ q,
                                                const bf16_t* __restrict__ k,
                                                const bf16_t* __restrict__ vt,
                                                bf16_t* __restrict__ y) {
  __shared__ __align__(16) bf16_t pls[8 * 16 * 40];  // per-wave 16x32 P tile, stride 40
  const int lane = threadIdx.x & 31;
  const int wid  = threadIdx.x >> 5;
  const int nlo  = lane & 15;
  const int hi   = lane >> 4;
  const int widx = blockIdx.x * 8 + wid;
  const int qt = widx & 127;          // T/16 = 128 q-tiles
  const int h  = (widx >> 7) & 15;
  const int bb = widx >> 11;
  const int i0 = qt * 16;
  const int kv = h >> 2;
  bf16_t* P = pls + wid * (16 * 40);

  // Q fragments: rows i0..i0+15, four K=32 chunks over head_dim
  v16bf aq[4];
  const bf16_t* qrow = q + (((long)bb * SEQ + i0 + nlo) * NH + h) * HD;
#pragma unroll
  for (int ks = 0; ks < 4; ++ks) aq[ks] = load_a(qrow + ks * 32, hi);

  v8f o[8];
#pragma unroll
  for (int dt = 0; dt < 8; ++dt) o[dt] = (v8f){0.f,0.f,0.f,0.f,0.f,0.f,0.f,0.f};
  float mrow[8], lrow[8];
#pragma unroll
  for (int r = 0; r < 8; ++r) { mrow[r] = -3.0e38f; lrow[r] = 0.f; }

  const int kthi  = (i0 + 15) >> 5;
  int lo = i0 - WIN; if (lo < 0) lo = 0;
  const int ktlo = lo >> 5;

  for (int kt = 0; kt <= kthi; ++kt) {
    const int j0 = kt * 32;
    // ----- S = Q K^T (two 16x16 fragments: keys j0..+15 and j0+16..+31) -----
    v8f s0 = (v8f){0.f,0.f,0.f,0.f,0.f,0.f,0.f,0.f};
    v8f s1 = s0;
    const bf16_t* kr0 = k + (((long)bb * SEQ + j0 + nlo) * NKV + kv) * HD;
    const bf16_t* kr1 = kr0 + (long)16 * NKV * HD;
#pragma unroll
    for (int ks = 0; ks < 4; ++ks) {
      v16bf b0 = load_b(kr0 + ks * 32, hi);
      v16bf b1 = load_b(kr1 + ks * 32, hi);
      s0 = wmma_bf16(aq[ks], b0, s0);
      s1 = wmma_bf16(aq[ks], b1, s1);
    }
    // ----- mask + online softmax -----
    float p0[8], p1[8];
#pragma unroll
    for (int r = 0; r < 8; ++r) {
      const int i  = i0 + r + 8 * hi;
      const int ja = j0 + nlo, jb = ja + 16;
      float sa = s0[r] * SCALE, sb = s1[r] * SCALE;
      bool okA = (ja <= i) && (((i - ja) <= WIN) || (ja < SINKN));
      bool okB = (jb <= i) && (((i - jb) <= WIN) || (jb < SINKN));
      sa = okA ? sa : -3.0e38f;
      sb = okB ? sb : -3.0e38f;
      float mx = fmaxf(sa, sb);
#pragma unroll
      for (int off = 1; off < 16; off <<= 1) mx = fmaxf(mx, __shfl_xor(mx, off, 32));
      const float mnew = fmaxf(mrow[r], mx);
      const float corr = __expf(mrow[r] - mnew);
      const float ea = __expf(sa - mnew);
      const float eb = __expf(sb - mnew);
      float sum = ea + eb;
#pragma unroll
      for (int off = 1; off < 16; off <<= 1) sum += __shfl_xor(sum, off, 32);
      lrow[r] = lrow[r] * corr + sum;
      mrow[r] = mnew;
      p0[r] = ea; p1[r] = eb;
#pragma unroll
      for (int dt = 0; dt < 8; ++dt) o[dt][r] *= corr;
    }
    // ----- relayout P (C-frag -> A-frag) through per-wave LDS tile -----
#pragma unroll
    for (int r = 0; r < 8; ++r) {
      const int m = r + 8 * hi;
      P[m * 40 + nlo]      = f2bf(p0[r]);
      P[m * 40 + 16 + nlo] = f2bf(p1[r]);
    }
    asm volatile("s_wait_dscnt 0" ::: "memory");
    const bf16_t* prow = P + nlo * 40;
    v16bf ap = load_a(prow, hi);
    // ----- O += P * V  (Vt rows are key-contiguous -> NT B fragments) -----
    const bf16_t* vrow = vt + (((long)bb * NKV + kv) * HD + nlo) * SEQ + j0;
#pragma unroll
    for (int dt = 0; dt < 8; ++dt) {
      v16bf bv = load_b(vrow + (long)dt * 16 * SEQ, hi);
      o[dt] = wmma_bf16(ap, bv, o[dt]);
    }
    // after the sink tile (kt==0), jump to the window start
    if (kt == 0 && ktlo > 1) kt = ktlo - 1;
  }

  // ----- normalize and store y (bf16) -----
#pragma unroll
  for (int r = 0; r < 8; ++r) {
    const float invl = 1.0f / lrow[r];
    const long yb = (((long)bb * SEQ + i0 + r + 8 * hi) * NH + h) * HD + nlo;
#pragma unroll
    for (int dt = 0; dt < 8; ++dt) y[yb + dt * 16] = f2bf(o[dt][r] * invl);
  }
}

// ---------- host orchestration ----------
extern "C" void kernel_launch(void* const* d_in, const int* in_sizes, int n_in,
                              void* d_out, int out_size, void* d_ws, size_t ws_size,
                              hipStream_t stream) {
  (void)in_sizes; (void)n_in; (void)out_size; (void)ws_size;
  const float* x  = (const float*)d_in[0];
  const float* wq = (const float*)d_in[1];
  const float* wk = (const float*)d_in[2];
  const float* wv = (const float*)d_in[3];
  const float* wo = (const float*)d_in[4];

  const long MT  = (long)BSZ * SEQ;        // 4096 rows
  const long KVC = (long)NKV * HD;         // 512

  char* ws = (char*)d_ws;
  size_t off = 0;
  auto carve = [&](size_t bytes) -> char* {
    char* p = ws + off;
    off += (bytes + 255) & ~(size_t)255;
    return p;
  };
  bf16_t* xb  = (bf16_t*)carve(MT * CDIM * 2);
  bf16_t* wqb = (bf16_t*)carve((long)CDIM * CDIM * 2);
  bf16_t* wkb = (bf16_t*)carve(KVC * CDIM * 2);
  bf16_t* wvb = (bf16_t*)carve(KVC * CDIM * 2);
  bf16_t* wob = (bf16_t*)carve((long)CDIM * CDIM * 2);
  float*  qf  = (float*) carve(MT * NH * HD * 4);
  float*  kf  = (float*) carve(MT * KVC * 4);
  float*  vf  = (float*) carve(MT * KVC * 4);
  bf16_t* qb  = (bf16_t*)carve(MT * NH * HD * 2);
  bf16_t* kb  = (bf16_t*)carve(MT * KVC * 2);
  bf16_t* vtb = (bf16_t*)carve(MT * KVC * 2);
  bf16_t* yb  = (bf16_t*)qf;  // reuse fp32-q region after RoPE consumed it

  auto cvt = [&](const float* in, bf16_t* outp, long n) {
    long n8 = n / 8;
    long blocks = (n8 + 255) / 256;
    if (blocks > 2048) blocks = 2048;
    convert_bf16_v8<<<dim3((unsigned)blocks), dim3(256), 0, stream>>>(in, outp, n8);
  };
  cvt(x,  xb,  MT * CDIM);
  cvt(wq, wqb, (long)CDIM * CDIM);
  cvt(wk, wkb, KVC * CDIM);
  cvt(wv, wvb, KVC * CDIM);
  cvt(wo, wob, (long)CDIM * CDIM);

  // projections  (block tile 256M x 128N)
  gemm_nt<<<dim3(CDIM / 128, (unsigned)(MT / 256)), 256, 0, stream>>>(
      xb, wqb, qf, (int)MT, CDIM, CDIM);
  gemm_nt<<<dim3((unsigned)(KVC / 128), (unsigned)(MT / 256)), 256, 0, stream>>>(
      xb, wkb, kf, (int)MT, (int)KVC, CDIM);
  gemm_nt<<<dim3((unsigned)(KVC / 128), (unsigned)(MT / 256)), 256, 0, stream>>>(
      xb, wvb, vf, (int)MT, (int)KVC, CDIM);

  // rope + bf16 cast
  long nq = MT * NH * 64;
  rope_convert<<<dim3((unsigned)((nq + 255) / 256)), 256, 0, stream>>>(qf, qb, NH, nq);
  long nk = MT * NKV * 64;
  rope_convert<<<dim3((unsigned)((nk + 255) / 256)), 256, 0, stream>>>(kf, kb, NKV, nk);

  // V transpose to (B,NKV,HD,T)
  transpose_v<<<dim3(HD / 32, SEQ / 32, BSZ * NKV), 256, 0, stream>>>(vf, vtb);

  // attention: B*NH*(T/16) waves, 8 waves per block
  attn_fwd<<<dim3((BSZ * NH * (SEQ / 16)) / 8), 256, 0, stream>>>(qb, kb, vtb, yb);

  // output projection -> fp32 d_out
  gemm_nt<<<dim3(CDIM / 128, (unsigned)(MT / 256)), 256, 0, stream>>>(
      yb, wob, (float*)d_out, (int)MT, CDIM, CDIM);
}
